// ContactCNN_36893769073428
// MI455X (gfx1250) — matrix operand order repaired and checked by reading, not compiled, not run on hardware
//
#include <hip/hip_runtime.h>
#include <hip/hip_bf16.h>

typedef __attribute__((ext_vector_type(16))) _Float16 v16h;
typedef __attribute__((ext_vector_type(8)))  _Float16 v8h;
typedef __attribute__((ext_vector_type(8)))  float    v8f;

#define HH   64
#define LL   512
#define NB   4
#define NPIX (NB*LL*LL)          /* 1048576 */
#define NROW (NB*LL)             /* 2048 rows per projected tensor */

static __device__ __forceinline__ v8h relu8(v8h x) {
    v8h z = {};
    return __builtin_elementwise_max(x, z);
}

// ---------------------------------------------------------------------------
// K1: p0 = z0·W^T ; p1 = z1·W^T + b_fc     (trivial FLOPs, plain VALU)
// ---------------------------------------------------------------------------
__global__ void k_proj(const float* __restrict__ z0, const float* __restrict__ z1,
                       const float* __restrict__ Wfc, const float* __restrict__ bfc,
                       float* __restrict__ p0, float* __restrict__ p1) {
    int t = blockIdx.x * 256 + threadIdx.x;     // 0 .. 262143
    int which = t >> 17;
    int rem   = t & 131071;                     // = row*64 + h
    int h     = rem & 63;
    int row   = rem >> 6;
    const float4* zr = (const float4*)((which ? z1 : z0) + row * 128);
    const float4* wr = (const float4*)(Wfc + h * 128);
    float acc = 0.f;
#pragma unroll
    for (int d = 0; d < 32; ++d) {
        float4 a = zr[d], w = wr[d];
        acc += a.x*w.x + a.y*w.y + a.z*w.z + a.w*w.w;
    }
    if (which) p1[rem] = acc + bfc[h];
    else       p0[rem] = acc;
}

// ---------------------------------------------------------------------------
// K2: BN1 stats per h (analytic mean/var of p0[b,i]+p1[b,j] incl. covariance
// over shared b), then pack u0 = a*p0, u1 = a*p1 + d as f16. One block per h.
// ---------------------------------------------------------------------------
__global__ void k_stats(const float* __restrict__ p0, const float* __restrict__ p1,
                        const float* __restrict__ g1, const float* __restrict__ be1,
                        _Float16* __restrict__ u0, _Float16* __restrict__ u1) {
    const int h = blockIdx.x, tid = threadIdx.x;
    float s0b[4] = {0,0,0,0}, s1b[4] = {0,0,0,0}, q0 = 0.f, q1 = 0.f;
    for (int e = tid; e < NROW; e += 256) {
        int b = e >> 9;
        float a = p0[e*HH + h]; s0b[b] += a; q0 += a*a;
        float c = p1[e*HH + h]; s1b[b] += c; q1 += c*c;
    }
    __shared__ float red[256];
    __shared__ float res[10];
    float vals[10] = {s0b[0],s0b[1],s0b[2],s0b[3],
                      s1b[0],s1b[1],s1b[2],s1b[3], q0, q1};
    for (int v = 0; v < 10; ++v) {
        red[tid] = vals[v]; __syncthreads();
        for (int off = 128; off > 0; off >>= 1) {
            if (tid < off) red[tid] += red[tid + off];
            __syncthreads();
        }
        if (tid == 0) res[v] = red[0];
        __syncthreads();
    }
    __shared__ float sA, sD;
    if (tid == 0) {
        float S0 = res[0]+res[1]+res[2]+res[3];
        float S1 = res[4]+res[5]+res[6]+res[7];
        float Mx = S0 * (1.f/2048.f), My = S1 * (1.f/2048.f);
        float Ex2 = res[8] * (1.f/2048.f), Ey2 = res[9] * (1.f/2048.f);
        float Exy = 0.f;
        for (int b = 0; b < 4; ++b)
            Exy += (res[b] * (1.f/512.f)) * (res[4+b] * (1.f/512.f));
        Exy *= 0.25f;
        float mean = Mx + My;
        float var  = (Ex2 - Mx*Mx) + (Ey2 - My*My) + 2.f*(Exy - Mx*My);
        float al = g1[h] * rsqrtf(var + 1e-5f);
        sA = al; sD = be1[h] - al * mean;
    }
    __syncthreads();
    float al = sA, de = sD;
    for (int e = tid; e < NROW; e += 256) {
        u0[e*HH + h] = (_Float16)(al * p0[e*HH + h]);
        u1[e*HH + h] = (_Float16)(al * p1[e*HH + h] + de);
    }
}

// ---------------------------------------------------------------------------
// K2b: pre-pack the WMMA B-fragment (conv weights, [K=64][N=16], taps 0..8,
// rest zero) into exact per-lane register order: wfrag[lane][e], e=0..31
// (first 16 = K 0..31 fragment, next 16 = K 32..63 fragment).
// B layout: lane L holds N = L%16; element e holds K = (e&7) + 8*(L/16) + 16*((e>>3)&1) (+32).
// ---------------------------------------------------------------------------
__global__ void k_packw(const float* __restrict__ convw, _Float16* __restrict__ wfrag) {
    int idx = blockIdx.x * 256 + threadIdx.x;   // 0..1023
    int lane = idx >> 5, e = idx & 31;
    int half = lane >> 4, n = lane & 15;
    int kh = (e & 7) + 8 * half + 16 * ((e >> 3) & 1) + 32 * (e >> 4);
    wfrag[lane * 32 + e] = (n < 9) ? (_Float16)convw[kh * 9 + n] : (_Float16)0.0f;
}

// ---------------------------------------------------------------------------
// K3: heavy pass, register-only fragment build (no LDS).
// Per wave-task: fixed (b,i), 64 consecutive j in 4 groups of 16.
// Lane (m = lane&15, half = lane>>4) builds its A fragment from four
// contiguous 8-half runs of relu(u0[b,i,:] + u1[b,j0+m,:]) per group:
//   a0 = h in [8h8, +8) ++ [16+8h8, +8),  a1 = same +32   (h8 = 8*half)
// Two v_wmma_f32_16x16x32_f16 accumulate K=64 -> D[16 pix][16 taps].
// Writes t[tap][b][i][j] planes (37.7 MB, L2-resident).
// ---------------------------------------------------------------------------
__global__ void k_conv(const _Float16* __restrict__ u0, const _Float16* __restrict__ u1,
                       const _Float16* __restrict__ wfrag, float* __restrict__ t) {
    const int tid  = threadIdx.x;
    const int wave = tid >> 5, lane = tid & 31;
    const int T  = blockIdx.x * 8 + wave;       // 16384 wave-tasks
    const int jq = T & 7;
    const int i  = (T >> 3) & 511;
    const int b  = T >> 12;
    const int m = lane & 15, half = lane >> 4;
    const int hb = 8 * half;

    // loop-invariant: u0 row runs + B fragments
    const _Float16* u0r = u0 + (b*LL + i) * HH;
    const v8h x0 = *(const v8h*)(u0r + hb);
    const v8h x1 = *(const v8h*)(u0r + 16 + hb);
    const v8h x2 = *(const v8h*)(u0r + 32 + hb);
    const v8h x3 = *(const v8h*)(u0r + 48 + hb);
    const v16h bf0 = *(const v16h*)(wfrag + lane * 32);
    const v16h bf1 = *(const v16h*)(wfrag + lane * 32 + 16);

#pragma unroll
    for (int g = 0; g < 4; ++g) {
        const int jrow = (jq << 6) + (g << 4) + m;
        const _Float16* u1r = u1 + (b*LL + jrow) * HH;
        v8h s0 = relu8(x0 + *(const v8h*)(u1r + hb));
        v8h s1 = relu8(x1 + *(const v8h*)(u1r + 16 + hb));
        v8h s2 = relu8(x2 + *(const v8h*)(u1r + 32 + hb));
        v8h s3 = relu8(x3 + *(const v8h*)(u1r + 48 + hb));
        v16h a0 = __builtin_shufflevector(s0, s1, 0,1,2,3,4,5,6,7,8,9,10,11,12,13,14,15);
        v16h a1 = __builtin_shufflevector(s2, s3, 0,1,2,3,4,5,6,7,8,9,10,11,12,13,14,15);

        v8f c = {};
        c = __builtin_amdgcn_wmma_f32_16x16x32_f16(false, a0, false, bf0, (short)0, c, false, false);
        c = __builtin_amdgcn_wmma_f32_16x16x32_f16(false, a1, false, bf1, (short)0, c, false, false);

        // D layout: lane holds tap n = m; rows M = r + 8*half = pixel j offset
        if (m < 9) {
            float* tp = t + ((((size_t)m*NB + b)*LL + i) << 9) + (jq << 6) + (g << 4) + hb;
#pragma unroll
            for (int r = 0; r < 8; ++r) tp[r] = c[r];
        }
    }
}

// ---------------------------------------------------------------------------
// K4: 9-tap stencil sum over t planes (SAME padding -> skip OOB), + conv_b,
// write pre-BN s to out, emit per-block (sum, sumsq) partials (deterministic).
// ---------------------------------------------------------------------------
__global__ void k_stencil(const float* __restrict__ t, const float* __restrict__ convb,
                          float* __restrict__ out, float* __restrict__ part) {
    int pix = blockIdx.x * 256 + threadIdx.x;
    int b = pix >> 18, rem = pix & 262143;
    int i = rem >> 9, j = rem & 511;
    float s = convb[0];
#pragma unroll
    for (int r = 0; r < 3; ++r) {
        int ii = i + r - 1;
        if (ii >= 0 && ii < 512) {
#pragma unroll
            for (int cc = 0; cc < 3; ++cc) {
                int jj = j + cc - 1;
                if (jj >= 0 && jj < 512) {
                    int k = r*3 + cc;
                    s += t[((((size_t)k*NB + b)*LL + ii) << 9) + jj];
                }
            }
        }
    }
    out[pix] = s;
    __shared__ float r1[256], r2[256];
    int tid = threadIdx.x;
    r1[tid] = s; r2[tid] = s*s; __syncthreads();
    for (int off = 128; off > 0; off >>= 1) {
        if (tid < off) { r1[tid] += r1[tid+off]; r2[tid] += r2[tid+off]; }
        __syncthreads();
    }
    if (tid == 0) { part[blockIdx.x] = r1[0]; part[4096 + blockIdx.x] = r2[0]; }
}

// K5: deterministic final reduction of 4096 block partials -> stats[0..1]
__global__ void k_reduce(const float* __restrict__ part, float* __restrict__ stats) {
    __shared__ float r1[256], r2[256];
    int tid = threadIdx.x;
    float s1 = 0.f, s2 = 0.f;
    for (int k = tid; k < 4096; k += 256) { s1 += part[k]; s2 += part[4096 + k]; }
    r1[tid] = s1; r2[tid] = s2; __syncthreads();
    for (int off = 128; off > 0; off >>= 1) {
        if (tid < off) { r1[tid] += r1[tid+off]; r2[tid] += r2[tid+off]; }
        __syncthreads();
    }
    if (tid == 0) { stats[0] = r1[0]; stats[1] = r2[0]; }
}

// K6: BN2 + sigmoid, in place on out
__global__ void k_final(float* __restrict__ out, const float* __restrict__ stats,
                        const float* __restrict__ g2, const float* __restrict__ b2) {
    int pix = blockIdx.x * 256 + threadIdx.x;
    float mean = stats[0] * (1.f / (float)NPIX);
    float var  = stats[1] * (1.f / (float)NPIX) - mean*mean;
    float y = (out[pix] - mean) * rsqrtf(var + 1e-5f) * g2[0] + b2[0];
    out[pix] = 1.f / (1.f + __expf(-y));
}

// ---------------------------------------------------------------------------
extern "C" void kernel_launch(void* const* d_in, const int* in_sizes, int n_in,
                              void* d_out, int out_size, void* d_ws, size_t ws_size,
                              hipStream_t stream) {
    const float* z0   = (const float*)d_in[0];
    const float* z1   = (const float*)d_in[1];
    const float* Wfc  = (const float*)d_in[2];
    const float* bfc  = (const float*)d_in[3];
    const float* g1   = (const float*)d_in[4];
    const float* be1  = (const float*)d_in[5];
    const float* cw   = (const float*)d_in[6];
    const float* cb   = (const float*)d_in[7];
    const float* g2   = (const float*)d_in[8];
    const float* b2   = (const float*)d_in[9];
    float* out = (float*)d_out;

    char* w = (char*)d_ws;
    float*    p0    = (float*)(w);                       // 512 KB
    float*    p1    = (float*)(w + 524288);              // 512 KB
    _Float16* u0    = (_Float16*)(w + 1048576);          // 256 KB
    _Float16* u1    = (_Float16*)(w + 1310720);          // 256 KB
    _Float16* wfrag = (_Float16*)(w + 1572864);          // 2 KB (B fragments)
    float*    tbuf  = (float*)(w + 1574912);             // 37.75 MB (L2 resident)
    float*    part  = (float*)(w + 39323648);            // 32 KB partials
    float*    stats = (float*)(w + 39356416);            // 8 B

    k_proj   <<<1024, 256, 0, stream>>>(z0, z1, Wfc, bfc, p0, p1);
    k_stats  <<<  64, 256, 0, stream>>>(p0, p1, g1, be1, u0, u1);
    k_packw  <<<   4, 256, 0, stream>>>(cw, wfrag);
    k_conv   <<<2048, 256, 0, stream>>>(u0, u1, wfrag, tbuf);
    k_stencil<<<4096, 256, 0, stream>>>(tbuf, cb, out, part);
    k_reduce <<<   1, 256, 0, stream>>>(part, stats);
    k_final  <<<4096, 256, 0, stream>>>(out, stats, g2, b2);
}
